// XnorBinaryLinear_44349832298614
// MI455X (gfx1250) — matrix operand order repaired and checked by reading, not compiled, not run on hardware
//
#include <hip/hip_runtime.h>
#include <stdint.h>

// ---------------------------------------------------------------------------
// XnorBinaryLinear on MI455X (gfx1250):
//   out = (x_bf16 @ sign(w - rowmean)_bf16^T) * scale_row + bias
// Compute-bound (AI ~1100 FLOP/B): V_WMMA_F32_16X16X32_BF16, f32 accumulate.
// Ternary weights are exact in bf16; only x is rounded (~2^-8 rel err).
// Global->LDS staging uses gfx1250 async loads (ASYNCcnt) so the double
// buffer costs zero VGPRs (round-1 version spilled its register staging).
// ---------------------------------------------------------------------------

typedef __bf16          v16bf __attribute__((ext_vector_type(16)));
typedef float           v8f   __attribute__((ext_vector_type(8)));
typedef unsigned short  v8u   __attribute__((ext_vector_type(8)));
typedef unsigned short  v16u  __attribute__((ext_vector_type(16)));

__device__ __forceinline__ unsigned short f2bf(float f) {
    // round-to-nearest-even f32 -> bf16 (bit trick)
    unsigned u = __float_as_uint(f);
    u += 0x7FFFu + ((u >> 16) & 1u);
    return (unsigned short)(u >> 16);
}

// low 32 bits of a flat shared pointer == workgroup-relative LDS byte offset
__device__ __forceinline__ unsigned lds_off(const void* p) {
    return (unsigned)(unsigned long long)(uintptr_t)p;
}

// gfx1250 async global->LDS copy, 16B per lane, tracked by ASYNCcnt
__device__ __forceinline__ void async_copy_b128(unsigned lds_dst, const void* gsrc) {
    asm volatile("global_load_async_to_lds_b128 %0, %1, off"
                 :: "v"(lds_dst), "v"(gsrc)
                 : "memory");
}

__device__ __forceinline__ void wait_async0() {
    asm volatile("s_wait_asynccnt 0" ::: "memory");
}

// ---------------- prep: x (f32) -> bf16 bits -------------------------------
__global__ void xnor_prep_x(const float* __restrict__ x,
                            unsigned short* __restrict__ xb,
                            long long total) {
    long long i = ((long long)blockIdx.x * blockDim.x + threadIdx.x) * 4;
    if (i >= total) return;
    float4 v = *(const float4*)(x + i);
    uint2 p;
    p.x = (unsigned)f2bf(v.x) | ((unsigned)f2bf(v.y) << 16);
    p.y = (unsigned)f2bf(v.z) | ((unsigned)f2bf(v.w) << 16);
    *(uint2*)(xb + i) = p;
}

// ---------------- prep: weight row -> sign (bf16 {-1,0,+1}) + scale --------
__global__ void xnor_prep_w(const float* __restrict__ w,
                            unsigned short* __restrict__ wsgn,
                            float* __restrict__ scale, int K) {
    const int row = blockIdx.x;
    const int t = threadIdx.x;
    const float* wr = w + (size_t)row * K;
    __shared__ float red[256];

    float s = 0.f;
    for (int k = t * 4; k < K; k += 256 * 4) {
        float4 v = *(const float4*)(wr + k);
        s += v.x + v.y + v.z + v.w;
    }
    red[t] = s; __syncthreads();
    for (int off = 128; off > 0; off >>= 1) {
        if (t < off) red[t] += red[t + off];
        __syncthreads();
    }
    const float mean = red[0] / (float)K;
    __syncthreads();

    float s2 = 0.f;
    for (int k = t * 4; k < K; k += 256 * 4) {
        float4 v = *(const float4*)(wr + k);
        s2 += fabsf(v.x - mean) + fabsf(v.y - mean) + fabsf(v.z - mean) + fabsf(v.w - mean);
    }
    red[t] = s2; __syncthreads();
    for (int off = 128; off > 0; off >>= 1) {
        if (t < off) red[t] += red[t + off];
        __syncthreads();
    }
    if (t == 0) scale[row] = red[0] / (float)K;

    unsigned short* dr = wsgn + (size_t)row * K;
    const unsigned short POS = 0x3F80u, NEG = 0xBF80u;  // bf16 +1 / -1
    for (int k = t * 4; k < K; k += 256 * 4) {
        float4 v = *(const float4*)(wr + k);
        unsigned short q0 = v.x > mean ? POS : (v.x < mean ? NEG : 0);
        unsigned short q1 = v.y > mean ? POS : (v.y < mean ? NEG : 0);
        unsigned short q2 = v.z > mean ? POS : (v.z < mean ? NEG : 0);
        unsigned short q3 = v.w > mean ? POS : (v.w < mean ? NEG : 0);
        uint2 p;
        p.x = (unsigned)q0 | ((unsigned)q1 << 16);
        p.y = (unsigned)q2 | ((unsigned)q3 << 16);
        *(uint2*)(dr + k) = p;
    }
}

// ---------------- main bf16 WMMA GEMM --------------------------------------
#define BM 128
#define BN 128
#define BK 64
#define KSTRIDE 72  // BK + 8 pad (elements); row stride 144B, 16B-aligned

__global__ __launch_bounds__(256) void xnor_gemm(
    const unsigned short* __restrict__ A,   // [M][K] bf16 bits (x)
    const unsigned short* __restrict__ Bw,  // [N][K] bf16 bits (sign)
    const float* __restrict__ scale,        // [N]
    const float* __restrict__ bias,         // [N]
    float* __restrict__ out,                // [M][N]
    int M, int N, int K) {
    __shared__ __align__(16) unsigned short As[2][BM * KSTRIDE];
    __shared__ __align__(16) unsigned short Bs[2][BN * KSTRIDE];

    const int tid = threadIdx.x;
    const int wid = tid >> 5;          // 8 waves (wave32)
    const int lane = tid & 31;
    const int laneHalf = lane >> 4;    // 0/1
    const int lanelo = lane & 15;

    const int bm = blockIdx.y * BM;
    const int bn = blockIdx.x * BN;
    const int wm = (wid & 1) * 64;     // wave M offset: 2 x 64
    const int wn = (wid >> 1) * 32;    // wave N offset: 4 x 32

    const unsigned short* gA = A + (size_t)bm * K;
    const unsigned short* gB = Bw + (size_t)bn * K;

    v8f acc[4][2];
    const v8f vzero = {0.f, 0.f, 0.f, 0.f, 0.f, 0.f, 0.f, 0.f};
#pragma unroll
    for (int i = 0; i < 4; ++i)
#pragma unroll
        for (int j = 0; j < 2; ++j) acc[i][j] = vzero;

    // tile fill geometry: 8 chunks of 8 elems (16B) per row, 128 rows,
    // 1024 chunks / 256 threads = 4 chunks each (per operand)
    const unsigned short* pA[4];
    const unsigned short* pB[4];
    unsigned lA[4], lB[4];
#pragma unroll
    for (int i = 0; i < 4; ++i) {
        int cid = tid + i * 256;
        int row = cid >> 3;
        int col = (cid & 7) * 8;
        pA[i] = gA + (size_t)row * K + col;
        pB[i] = gB + (size_t)row * K + col;
        lA[i] = (unsigned)(row * KSTRIDE + col) * 2u;  // byte offset in buffer
        lB[i] = lA[i];
    }
    const unsigned aBase0 = lds_off(&As[0][0]);
    const unsigned aBase1 = lds_off(&As[1][0]);
    const unsigned bBase0 = lds_off(&Bs[0][0]);
    const unsigned bBase1 = lds_off(&Bs[1][0]);

    // prologue: async-fill stage 0
#pragma unroll
    for (int i = 0; i < 4; ++i) {
        async_copy_b128(aBase0 + lA[i], pA[i]);
        async_copy_b128(bBase0 + lB[i], pB[i]);
        pA[i] += BK;
        pB[i] += BK;
    }
    wait_async0();
    __syncthreads();

    const int NK = K / BK;
    int stage = 0;
    for (int kt = 0; kt < NK; ++kt) {
        const bool more = (kt + 1) < NK;
        if (more) {
            // async-fill the other buffer; zero VGPR staging, overlaps compute
            const unsigned aB = stage ? aBase0 : aBase1;
            const unsigned bB = stage ? bBase0 : bBase1;
#pragma unroll
            for (int i = 0; i < 4; ++i) {
                async_copy_b128(aB + lA[i], pA[i]);
                async_copy_b128(bB + lB[i], pB[i]);
                pA[i] += BK;
                pB[i] += BK;
            }
        }

        // two K=32 WMMA steps out of the staged K=64 tile
#pragma unroll
        for (int ks = 0; ks < 2; ++ks) {
            const int kfA = ks * 32 + laneHalf * 8;   // A: K runs {kb..+7, kb+16..+23}
            const int kfB = ks * 32 + laneHalf * 8;   // B: K runs {kb..+7, kb+16..+23}

            v16bf af[4];
#pragma unroll
            for (int tm = 0; tm < 4; ++tm) {
                const unsigned short* p =
                    &As[stage][(wm + tm * 16 + lanelo) * KSTRIDE + kfA];
                v8u lo = *(const v8u*)(p);        // K = kb .. kb+7
                v8u hi = *(const v8u*)(p + 16);   // K = kb+16 .. kb+23
                v16u au = __builtin_shufflevector(lo, hi, 0, 1, 2, 3, 4, 5, 6, 7,
                                                  8, 9, 10, 11, 12, 13, 14, 15);
                __builtin_memcpy(&af[tm], &au, sizeof(au));
            }
            v16bf bfr[2];
#pragma unroll
            for (int tn = 0; tn < 2; ++tn) {
                const unsigned short* p =
                    &Bs[stage][(wn + tn * 16 + lanelo) * KSTRIDE + kfB];
                v8u lo = *(const v8u*)(p);
                v8u hi = *(const v8u*)(p + 16);
                v16u bu = __builtin_shufflevector(lo, hi, 0, 1, 2, 3, 4, 5, 6, 7,
                                                  8, 9, 10, 11, 12, 13, 14, 15);
                __builtin_memcpy(&bfr[tn], &bu, sizeof(bu));
            }
#pragma unroll
            for (int tm = 0; tm < 4; ++tm)
#pragma unroll
                for (int tn = 0; tn < 2; ++tn)
                    acc[tm][tn] = __builtin_amdgcn_wmma_f32_16x16x32_bf16(
                        false, af[tm], false, bfr[tn], (short)0, acc[tm][tn],
                        false, false);
        }

        if (more) {
            wait_async0();      // my async fills of the other buffer are done
            __syncthreads();    // everyone's are done; stage buffer consumed
            stage ^= 1;
        }
    }

    // epilogue: D layout — VGPR r: M = r + 8*(lane>=16), N = lane%16
#pragma unroll
    for (int tn = 0; tn < 2; ++tn) {
        const int col = bn + wn + tn * 16 + lanelo;
        const float sc = scale[col];
        const float bi = bias[col];
#pragma unroll
        for (int tm = 0; tm < 4; ++tm) {
            const int row0 = bm + wm + tm * 16 + laneHalf * 8;
#pragma unroll
            for (int r = 0; r < 8; ++r) {
                out[(size_t)(row0 + r) * N + col] = acc[tm][tn][r] * sc + bi;
            }
        }
    }
}

// ---------------------------------------------------------------------------
extern "C" void kernel_launch(void* const* d_in, const int* in_sizes, int n_in,
                              void* d_out, int out_size, void* d_ws, size_t ws_size,
                              hipStream_t stream) {
    const float* x = (const float*)d_in[0];
    const float* w = (const float*)d_in[1];
    const float* bias = (const float*)d_in[2];
    float* out = (float*)d_out;

    const int OUT = in_sizes[2];                       // 11008
    const int K = (int)((long long)in_sizes[1] / OUT); // 4096
    const int M = (int)((long long)in_sizes[0] / K);   // 8192 (B*S)

    // workspace: xb [M*K] bf16 | wsgn [OUT*K] bf16 | scale [OUT] f32  (~157MB)
    unsigned short* xb = (unsigned short*)d_ws;
    unsigned short* wsgn = xb + (size_t)M * K;
    float* scale = (float*)(wsgn + (size_t)OUT * K);

    const long long totalx = (long long)M * K;
    const int blocksx = (int)((totalx / 4 + 255) / 256);
    xnor_prep_x<<<blocksx, 256, 0, stream>>>(x, xb, totalx);
    xnor_prep_w<<<OUT, 256, 0, stream>>>(w, wsgn, scale, K);

    dim3 grid(OUT / BN, M / BM);  // (86, 64)
    xnor_gemm<<<grid, 256, 0, stream>>>(xb, wsgn, scale, bias, out, M, OUT, K);
}
